// NGLOD_46780783788465
// MI455X (gfx1250) — matrix operand order repaired and checked by reading, not compiled, not run on hardware
//
#include <hip/hip_runtime.h>
#include <hip/hip_bf16.h>

typedef __attribute__((ext_vector_type(16))) _Float16 v16h;
typedef __attribute__((ext_vector_type(8)))  _Float16 v8h;
typedef __attribute__((ext_vector_type(8)))  float    v8f;

#define FSTR 40   // feats row stride in halves (80B, multiple of 16B)
#define XSTR 72   // activation row stride in halves (144B, multiple of 16B)

// ---------------------------------------------------------------------------
// Kernel 1: repack f32 weights into f16 WMMA B-fragment layout in d_ws.
// Fragment f, lane l, element j holds W[k][n] with
//   n = nt*16 + (l&15),  k = kt*32 + ((l>>4)<<4) + j   (zero-padded OOB)
// Fragment table (fixed order):
//   0..3   sW1 [32,64]  kt=0,      nt=f
//   4..5   sW2 [64,16]  kt=f-4,    nt=0
//   6..13  cW1 [43,64]  kt=(f-6)/4,nt=(f-6)%4   (K padded 43->64)
//   14..21 cW2 [64,64]  kt=(f-14)/4,nt=(f-14)%4
//   22..23 cW3 [64,3]   kt=f-22,   nt=0         (N padded 3->16)
// ---------------------------------------------------------------------------
__global__ void ngl_repack(const float* __restrict__ sW1, const float* __restrict__ sW2,
                           const float* __restrict__ cW1, const float* __restrict__ cW2,
                           const float* __restrict__ cW3, _Float16* __restrict__ frags) {
    int f = blockIdx.x;
    int lane = threadIdx.x;  // 32
    const float* W; int kt, nt, Kact, Nact, ld;
    if (f < 4)       { W = sW1; kt = 0;         nt = f;         Kact = 32; Nact = 64; ld = 64; }
    else if (f < 6)  { W = sW2; kt = f - 4;     nt = 0;         Kact = 64; Nact = 16; ld = 16; }
    else if (f < 14) { W = cW1; kt = (f-6)/4;   nt = (f-6)&3;   Kact = 43; Nact = 64; ld = 64; }
    else if (f < 22) { W = cW2; kt = (f-14)/4;  nt = (f-14)&3;  Kact = 64; Nact = 64; ld = 64; }
    else             { W = cW3; kt = f - 22;    nt = 0;         Kact = 64; Nact = 3;  ld = 3;  }
    int n = nt * 16 + (lane & 15);
    int kbase = kt * 32 + ((lane >> 4) << 4);
    _Float16* dst = frags + ((size_t)f * 32 + lane) * 16;
    #pragma unroll
    for (int j = 0; j < 16; ++j) {
        int k = kbase + j;
        float v = (k < Kact && n < Nact) ? W[(size_t)k * ld + n] : 0.0f;
        dst[j] = (_Float16)v;
    }
}

// ---------------------------------------------------------------------------
// Helpers for the main kernel
// ---------------------------------------------------------------------------
__device__ __forceinline__ v16h cat8(v8h lo, v8h hi) {
    v16h r;
    #pragma unroll
    for (int i = 0; i < 8; ++i) { r[i] = lo[i]; r[i + 8] = hi[i]; }
    return r;
}

// A-fragment (16-bit 16x32 layout): lane row = caller's row pointer; chunks:
// halves [kt*32 + hi*16, +8) and [kt*32 + hi*16 + 8, +8)
__device__ __forceinline__ v16h load_a(const _Float16* row, int kt, int hi) {
    v8h lo = *(const v8h*)(row + kt * 32 + hi * 16);
    v8h hh = *(const v8h*)(row + kt * 32 + hi * 16 + 8);
    return cat8(lo, hh);
}

__device__ __forceinline__ v16h load_b(const _Float16* __restrict__ frags, int fid, int lane) {
    const v8h* p = (const v8h*)(frags + ((size_t)fid * 32 + lane) * 16);
    return cat8(p[0], p[1]);
}

__device__ __forceinline__ v8f wmma_f16(v16h a, v16h b, v8f c) {
    return __builtin_amdgcn_wmma_f32_16x16x32_f16(false, a, false, b, (short)0, c, false, false);
}

// ---------------------------------------------------------------------------
// Kernel 2: main fused kernel. 256 threads = 8 waves = 128 points / block.
// ---------------------------------------------------------------------------
__global__ void __launch_bounds__(256)
ngl_main(const float* __restrict__ pts, const float* __restrict__ dirs,
         const float* __restrict__ g0, const float* __restrict__ g1,
         const float* __restrict__ g2, const float* __restrict__ g3,
         const _Float16* __restrict__ frags,
         const float* __restrict__ sB1, const float* __restrict__ sB2,
         const float* __restrict__ cB1, const float* __restrict__ cB2,
         const float* __restrict__ cB3,
         float* __restrict__ out_color, float* __restrict__ out_sigma, int NP) {
    __shared__ __align__(16) _Float16 feats[128 * FSTR];  // 32 trilerp features / point
    __shared__ __align__(16) _Float16 xbuf[128 * XSTR];   // x = [h(16) | pe(27) | 0-pad]
    __shared__ __align__(16) _Float16 hbuf[128 * XSTR];   // inter-layer staging
    __shared__ int maskv[128];

    const int tid = threadIdx.x;
    const int blockBase = blockIdx.x * 128;

    // ---------------- Phase A ----------------
    if (tid < 128) {
        int p = blockBase + tid;
        int pc = p < NP ? p : NP - 1;
        const float inv3 = 1.0f / 3.0f;
        float px = pts[(size_t)pc * 3 + 0] * inv3;
        float py = pts[(size_t)pc * 3 + 1] * inv3;
        float pz = pts[(size_t)pc * 3 + 2] * inv3;
        maskv[tid] = (fabsf(px) < 0.5f && fabsf(py) < 0.5f && fabsf(pz) < 0.5f) ? 1 : 0;
        px = fminf(fmaxf(px + 0.5f, 0.0f), 1.0f);
        py = fminf(fmaxf(py + 0.5f, 0.0f), 1.0f);
        pz = fminf(fmaxf(pz + 0.5f, 0.0f), 1.0f);

        float acc[32];
        #pragma unroll
        for (int i = 0; i < 32; ++i) acc[i] = 0.0f;

        const float* grids[4] = { g0, g1, g2, g3 };
        const int ress[4] = { 16, 32, 64, 128 };
        #pragma unroll
        for (int lod = 0; lod < 4; ++lod) {
            const int r = ress[lod];
            const float* g = grids[lod];
            float xs = px * (r - 1), ys = py * (r - 1), zs = pz * (r - 1);
            int cx = (int)floorf(xs); cx = cx < 0 ? 0 : (cx > r - 2 ? r - 2 : cx);
            int cy = (int)floorf(ys); cy = cy < 0 ? 0 : (cy > r - 2 ? r - 2 : cy);
            int cz = (int)floorf(zs); cz = cz < 0 ? 0 : (cz > r - 2 ? r - 2 : cz);
            float fx = xs - cx, fy = ys - cy, fz = zs - cz;
            float wx[2] = { 1.0f - fx, fx };
            float wy[2] = { 1.0f - fy, fy };
            float wz[2] = { 1.0f - fz, fz };
            #pragma unroll
            for (int i = 0; i < 2; ++i)
            #pragma unroll
            for (int j = 0; j < 2; ++j)
            #pragma unroll
            for (int k = 0; k < 2; ++k) {
                float w = wx[i] * wy[j] * wz[k];
                size_t flat = ((size_t)((cx + i) * r + (cy + j)) * r + (cz + k));
                const float4* rp = (const float4*)(g + flat * 8);
                float4 r0 = rp[0], r1 = rp[1];
                acc[0  + lod] += w * r0.x; acc[4  + lod] += w * r0.y;
                acc[8  + lod] += w * r0.z; acc[12 + lod] += w * r0.w;
                acc[16 + lod] += w * r1.x; acc[20 + lod] += w * r1.y;
                acc[24 + lod] += w * r1.z; acc[28 + lod] += w * r1.w;
            }
        }
        _Float16* fr = feats + tid * FSTR;
        #pragma unroll
        for (int i = 0; i < 32; ++i) fr[i] = (_Float16)acc[i];
    } else {
        int m = tid - 128;
        int p = blockBase + m;
        int pc = p < NP ? p : NP - 1;
        float dx = dirs[(size_t)pc * 3 + 0];
        float dy = dirs[(size_t)pc * 3 + 1];
        float dz = dirs[(size_t)pc * 3 + 2];
        _Float16* xr = xbuf + m * XSTR;
        xr[16] = (_Float16)dx; xr[17] = (_Float16)dy; xr[18] = (_Float16)dz;
        #pragma unroll
        for (int j = 0; j < 4; ++j) {
            float sc = (float)(1 << j);
            xr[19 + 6 * j] = (_Float16)__sinf(sc * dx);
            xr[20 + 6 * j] = (_Float16)__sinf(sc * dy);
            xr[21 + 6 * j] = (_Float16)__sinf(sc * dz);
            xr[22 + 6 * j] = (_Float16)__cosf(sc * dx);
            xr[23 + 6 * j] = (_Float16)__cosf(sc * dy);
            xr[24 + 6 * j] = (_Float16)__cosf(sc * dz);
        }
        #pragma unroll
        for (int c = 43; c < 64; ++c) xr[c] = (_Float16)0.0f;  // K zero-pad
    }
    __syncthreads();

    // ---------------- Phase B: 24 WMMAs per wave (16 points) ----------------
    const int wv   = tid >> 5;
    const int lane = tid & 31;
    const int col  = lane & 15;
    const int hi   = lane >> 4;
    const int rowM = wv * 16 + 8 * hi;  // + r gives block-local point of C-row r

    const _Float16* farow = feats + (wv * 16 + col) * FSTR;
    const _Float16* xrow  = xbuf  + (wv * 16 + col) * XSTR;
    const _Float16* hrow  = hbuf  + (wv * 16 + col) * XSTR;

    // Layer 1: feats[16,32] @ sW1[32,64] + sB1, relu -> hbuf
    {
        v16h a = load_a(farow, 0, hi);
        #pragma unroll
        for (int nt = 0; nt < 4; ++nt) {
            v8f acc = {};
            acc = wmma_f16(a, load_b(frags, nt, lane), acc);
            float bias = sB1[nt * 16 + col];
            #pragma unroll
            for (int r = 0; r < 8; ++r) {
                float v = fmaxf(acc[r] + bias, 0.0f);
                hbuf[(rowM + r) * XSTR + nt * 16 + col] = (_Float16)v;
            }
        }
    }
    // Layer 2: h1[16,64] @ sW2[64,16] + sB2 -> x[:,0:16]; keep col 0 for sigma
    float hsig[8];
    {
        v16h a0 = load_a(hrow, 0, hi);
        v16h a1 = load_a(hrow, 1, hi);
        v8f acc = {};
        acc = wmma_f16(a0, load_b(frags, 4, lane), acc);
        acc = wmma_f16(a1, load_b(frags, 5, lane), acc);
        float bias = sB2[col];
        #pragma unroll
        for (int r = 0; r < 8; ++r) {
            float v = acc[r] + bias;
            hsig[r] = v;
            xbuf[(rowM + r) * XSTR + col] = (_Float16)v;
        }
    }
    // Layer 3: x[16,64(43)] @ cW1 + cB1, relu -> hbuf
    {
        v16h a0 = load_a(xrow, 0, hi);
        v16h a1 = load_a(xrow, 1, hi);
        #pragma unroll
        for (int nt = 0; nt < 4; ++nt) {
            v8f acc = {};
            acc = wmma_f16(a0, load_b(frags, 6 + nt, lane), acc);
            acc = wmma_f16(a1, load_b(frags, 10 + nt, lane), acc);
            float bias = cB1[nt * 16 + col];
            #pragma unroll
            for (int r = 0; r < 8; ++r) {
                float v = fmaxf(acc[r] + bias, 0.0f);
                hbuf[(rowM + r) * XSTR + nt * 16 + col] = (_Float16)v;
            }
        }
    }
    // Layer 4: c1[16,64] @ cW2 + cB2, relu -> hbuf (A loaded before stores)
    {
        v16h a0 = load_a(hrow, 0, hi);
        v16h a1 = load_a(hrow, 1, hi);
        #pragma unroll
        for (int nt = 0; nt < 4; ++nt) {
            v8f acc = {};
            acc = wmma_f16(a0, load_b(frags, 14 + nt, lane), acc);
            acc = wmma_f16(a1, load_b(frags, 18 + nt, lane), acc);
            float bias = cB2[nt * 16 + col];
            #pragma unroll
            for (int r = 0; r < 8; ++r) {
                float v = fmaxf(acc[r] + bias, 0.0f);
                hbuf[(rowM + r) * XSTR + nt * 16 + col] = (_Float16)v;
            }
        }
    }
    // Layer 5: c2[16,64] @ cW3[64,16(3)]
    v8f acc5 = {};
    {
        v16h a0 = load_a(hrow, 0, hi);
        v16h a1 = load_a(hrow, 1, hi);
        acc5 = wmma_f16(a0, load_b(frags, 22, lane), acc5);
        acc5 = wmma_f16(a1, load_b(frags, 23, lane), acc5);
    }
    // Epilogue: sigmoid + mask for color, exp + mask for sigma
    if (col < 3) {
        float bias = cB3[col];
        #pragma unroll
        for (int r = 0; r < 8; ++r) {
            int pl = rowM + r;
            int p = blockBase + pl;
            if (p < NP) {
                float v = 1.0f / (1.0f + __expf(-(acc5[r] + bias)));
                out_color[(size_t)p * 3 + col] = maskv[pl] ? v : 0.0f;
            }
        }
    }
    if (col == 0) {
        #pragma unroll
        for (int r = 0; r < 8; ++r) {
            int pl = rowM + r;
            int p = blockBase + pl;
            if (p < NP)
                out_sigma[p] = maskv[pl] ? __expf(hsig[r]) : 0.0f;
        }
    }
}

extern "C" void kernel_launch(void* const* d_in, const int* in_sizes, int n_in,
                              void* d_out, int out_size, void* d_ws, size_t ws_size,
                              hipStream_t stream) {
    const float* pts  = (const float*)d_in[0];
    const float* dirs = (const float*)d_in[1];
    const float* g0   = (const float*)d_in[2];
    const float* g1   = (const float*)d_in[3];
    const float* g2   = (const float*)d_in[4];
    const float* g3   = (const float*)d_in[5];
    const float* sW1  = (const float*)d_in[6];
    const float* sB1  = (const float*)d_in[7];
    const float* sW2  = (const float*)d_in[8];
    const float* sB2  = (const float*)d_in[9];
    const float* cW1  = (const float*)d_in[10];
    const float* cB1  = (const float*)d_in[11];
    const float* cW2  = (const float*)d_in[12];
    const float* cB2  = (const float*)d_in[13];
    const float* cW3  = (const float*)d_in[14];
    const float* cB3  = (const float*)d_in[15];

    int NP = in_sizes[0] / 3;  // 524288
    _Float16* frags = (_Float16*)d_ws;  // 24 frags * 512 halves = 24 KB
    float* out_color = (float*)d_out;
    float* out_sigma = out_color + (size_t)NP * 3;

    ngl_repack<<<dim3(24), dim3(32), 0, stream>>>(sW1, sW2, cW1, cW2, cW3, frags);
    int nBlocks = (NP + 127) / 128;
    ngl_main<<<dim3(nBlocks), dim3(256), 0, stream>>>(
        pts, dirs, g0, g1, g2, g3, frags, sB1, sB2, cB1, cB2, cB3,
        out_color, out_sigma, NP);
}
